// LL_custom_76647986364883
// MI455X (gfx1250) — matrix operand order repaired and checked by reading, not compiled, not run on hardware
//
#include <hip/hip_runtime.h>
#include <hip/hip_bf16.h>

// ---------------- problem constants ----------------
#define BATCH   64
#define CIN     16
#define HIMG    32
#define FOLD    16
#define OCH     16
#define NCOL    4096            // OC*FOLD*FOLD
#define KTOT    36864           // CIN*FOLD*FOLD*3*3
#define SPLITK  16
#define KPS     (KTOT / SPLITK) // 2304
#define KSTEPS  (KPS / 32)      // 72
#define NKSTEP  (KTOT / 32)     // 1152
#define EPSBN   1e-5f

typedef __attribute__((ext_vector_type(16))) __bf16 v16bf;
typedef __attribute__((ext_vector_type(8)))  float  v8f;

// Native converts: clang lowers float<->__bf16 to gfx1250 v_cvt (RNE default).
__device__ __forceinline__ __bf16 to_bf16(float f)   { return (__bf16)f; }
__device__ __forceinline__ float  from_bf16(__bf16 h){ return (float)h; }

// ---------------------------------------------------------------------------
// Kernel 1: gather im2col patches and store them in a contiguous-K WMMA
// 16-bit A-fragment order (K = 16*(lane/16) + e), split into bf16 hi + lo.
// Storage element t = ((kstep*4 + mtile)*32 + lane)*16 + e
//   lanes 0-15 : M = lane,    e=0..15 -> K=e      (0..15)
//   lanes16-31 : M = lane-16, e=0..15 -> K=16+e   (16..31)
// ---------------------------------------------------------------------------
#define PACK_TOTAL (NKSTEP * 4 * 32 * 16)   // 2,359,296

__global__ void pack_patches_kernel(const float* __restrict__ x,
                                    __bf16* __restrict__ Ahi,
                                    __bf16* __restrict__ Alo) {
    int t = blockIdx.x * blockDim.x + threadIdx.x;
    if (t >= PACK_TOTAL) return;
    int e     = t & 15;
    int lane  = (t >> 4) & 31;
    int frag  = t >> 9;
    int mt    = frag & 3;
    int kstep = frag >> 2;

    int K = e + ((lane >= 16) ? 16 : 0);
    int i = kstep * 32 + K;              // flattened (c,fh,fw,kh,kw) index
    int b = mt * 16 + (lane & 15);       // batch row (M)

    int patch = i / 9;
    int r     = i - patch * 9;
    int kh = r / 3, kw = r - kh * 3;
    int c  = patch >> 8;
    int fh = (patch >> 4) & 15;
    int fw = patch & 15;
    int h  = fh * 2 + kh - 1;            // stride 2, pad 1
    int w  = fw * 2 + kw - 1;

    float v = 0.0f;
    if ((unsigned)h < (unsigned)HIMG && (unsigned)w < (unsigned)HIMG)
        v = x[(((b * CIN + c) << 5) + h) * 32 + w];

    __bf16 hi = to_bf16(v);
    __bf16 lo = to_bf16(v - from_bf16(hi));
    Ahi[t] = hi;
    Alo[t] = lo;
}

// ---------------------------------------------------------------------------
// Kernel 2: WMMA GEMM.  One wave per (N-pair, K-split).
//   M = 64 (4 tiles of 16), N-pair = 2 tiles of 16 columns, K slice = 2304.
// W (fp32, HBM-streamed once, 604 MB -> ~26 us at 23.3 TB/s) is converted to
// bf16 hi/lo in registers with native v_cvt; the per-lane load (16 consecutive
// K at column o, halves split by lane>=16) uses the same contiguous K order as the packed A-fragment
// layout.  bf16x3: hi*hi + lo*hi + hi*lo with fp32 accumulation.
// ---------------------------------------------------------------------------
__device__ __forceinline__ void cvt16_hilo(const float* __restrict__ p,
                                           v16bf& hi, v16bf& lo) {
    const float4* p4 = reinterpret_cast<const float4*>(p);
#pragma unroll
    for (int q = 0; q < 4; ++q) {
        float4 f = p4[q];
        float vals[4] = {f.x, f.y, f.z, f.w};
#pragma unroll
        for (int j = 0; j < 4; ++j) {
            __bf16 h = to_bf16(vals[j]);
            hi[q * 4 + j] = h;
            lo[q * 4 + j] = to_bf16(vals[j] - from_bf16(h));
        }
    }
}

__device__ __forceinline__ v8f wmma_bf16(v16bf a, v16bf b, v8f c) {
    return __builtin_amdgcn_wmma_f32_16x16x32_bf16(
        /*neg_a=*/false, a, /*neg_b=*/false, b,
        /*c_mod=*/(short)0, c, /*reuse_a=*/false, /*reuse_b=*/false);
}

__global__ void __launch_bounds__(32)
gemm_wmma_kernel(const float* __restrict__ W,
                 const v16bf* __restrict__ Ahi,
                 const v16bf* __restrict__ Alo,
                 float* __restrict__ out_part) {
    const int lane  = threadIdx.x;       // 0..31, uniform full-wave EXEC
    const int npair = blockIdx.x;        // 0..127  -> columns [npair*32, +32)
    const int ks    = blockIdx.y;        // 0..SPLITK-1
    const int kOff  = ks * KPS;

    const int col   = lane & 15;
    const int khalf = (lane >> 4) << 4;  // 0 or 16 (B-fragment K half)

    const float* w0 = W + (size_t)(npair * 32 + col) * KTOT + kOff + khalf;
    const float* w1 = w0 + (size_t)16 * KTOT;

    const v16bf* Ah = Ahi + lane;        // fragment base for this lane
    const v16bf* Al = Alo + lane;
    const int kg0 = ks * KSTEPS;         // first global kstep of this slice

    v8f acc[4][2];
#pragma unroll
    for (int mt = 0; mt < 4; ++mt) { acc[mt][0] = (v8f)0.0f; acc[mt][1] = (v8f)0.0f; }

    for (int s = 0; s < KSTEPS; ++s) {
        v16bf bhi0, blo0, bhi1, blo1;
        cvt16_hilo(w0 + s * 32, bhi0, blo0);
        cvt16_hilo(w1 + s * 32, bhi1, blo1);
        const int fb = (kg0 + s) * 128;  // (kstep*4 + mt)*32 fragment index
#pragma unroll
        for (int mt = 0; mt < 4; ++mt) {
            v16bf ah = Ah[fb + mt * 32];
            v16bf al = Al[fb + mt * 32];
            // bf16x3: hi*hi + lo*hi + hi*lo  (fp32 accumulate)
            acc[mt][0] = wmma_bf16(ah, bhi0, acc[mt][0]);
            acc[mt][0] = wmma_bf16(al, bhi0, acc[mt][0]);
            acc[mt][0] = wmma_bf16(ah, blo0, acc[mt][0]);
            acc[mt][1] = wmma_bf16(ah, bhi1, acc[mt][1]);
            acc[mt][1] = wmma_bf16(al, bhi1, acc[mt][1]);
            acc[mt][1] = wmma_bf16(ah, blo1, acc[mt][1]);
        }
    }

    // C/D layout: VGPR r -> M = r (lanes 0-15) / r+8 (lanes 16-31), N = lane&15
    const int o0   = npair * 32 + col;
    const int mAdd = (lane >> 4) << 3;
#pragma unroll
    for (int mt = 0; mt < 4; ++mt) {
#pragma unroll
        for (int r = 0; r < 8; ++r) {
            int b = mt * 16 + r + mAdd;
            size_t idx = ((size_t)(ks * BATCH + b) << 12) + o0;
            out_part[idx]      = acc[mt][0][r];
            out_part[idx + 16] = acc[mt][1][r];
        }
    }
}

// ---------------------------------------------------------------------------
// Kernel 3: per-channel BN statistics (training mode, biased variance).
// One block per OC channel; fixed-order split-K sum -> deterministic.
// ---------------------------------------------------------------------------
__global__ void bn_stats_kernel(const float* __restrict__ out_part,
                                const float* __restrict__ bias,
                                const float* __restrict__ gamma,
                                const float* __restrict__ beta,
                                float* __restrict__ scaleshift) {
    const int oc  = blockIdx.x;
    const int tid = threadIdx.x;
    float sum = 0.0f, sq = 0.0f;
    for (int idx = tid; idx < BATCH * 256; idx += 256) {
        int b = idx >> 8;
        int o = oc * 256 + (idx & 255);
        float v = bias[o];
#pragma unroll
        for (int p = 0; p < SPLITK; ++p)
            v += out_part[((size_t)(p * BATCH + b) << 12) + o];
        sum += v;
        sq  += v * v;
    }
    __shared__ float ssum[256], ssq[256];
    ssum[tid] = sum; ssq[tid] = sq;
    __syncthreads();
    for (int st = 128; st > 0; st >>= 1) {
        if (tid < st) { ssum[tid] += ssum[tid + st]; ssq[tid] += ssq[tid + st]; }
        __syncthreads();
    }
    if (tid == 0) {
        const float inv = 1.0f / (BATCH * 256.0f);
        float mean = ssum[0] * inv;
        float var  = ssq[0] * inv - mean * mean;
        float sc   = gamma[oc] * rsqrtf(var + EPSBN);
        scaleshift[oc]       = sc;
        scaleshift[16 + oc]  = beta[oc] - mean * sc;
    }
}

// ---------------------------------------------------------------------------
// Kernel 4: bias + BN affine + ReLU -> d_out [B, OC, FOLD, FOLD] fp32.
// ---------------------------------------------------------------------------
__global__ void bn_apply_kernel(const float* __restrict__ out_part,
                                const float* __restrict__ bias,
                                const float* __restrict__ scaleshift,
                                float* __restrict__ out) {
    int idx = blockIdx.x * blockDim.x + threadIdx.x;
    if (idx >= BATCH * NCOL) return;
    int b  = idx >> 12;
    int o  = idx & 4095;
    int oc = o >> 8;
    float v = bias[o];
#pragma unroll
    for (int p = 0; p < SPLITK; ++p)
        v += out_part[((size_t)(p * BATCH + b) << 12) + o];
    out[idx] = fmaxf(v * scaleshift[oc] + scaleshift[16 + oc], 0.0f);
}

// ---------------------------------------------------------------------------
extern "C" void kernel_launch(void* const* d_in, const int* in_sizes, int n_in,
                              void* d_out, int out_size, void* d_ws, size_t ws_size,
                              hipStream_t stream) {
    const float* x     = (const float*)d_in[0];
    const float* W     = (const float*)d_in[1];
    const float* bias  = (const float*)d_in[2];
    const float* gamma = (const float*)d_in[3];
    const float* beta  = (const float*)d_in[4];
    float* out = (float*)d_out;

    char* ws = (char*)d_ws;
    const size_t AFRAG_BYTES = (size_t)PACK_TOTAL * 2;          // 4,718,592
    __bf16* Ahi = (__bf16*)ws;
    __bf16* Alo = (__bf16*)(ws + AFRAG_BYTES);
    float* out_part   = (float*)(ws + 2 * AFRAG_BYTES);         // SPLITK*64*4096 f32
    float* scaleshift = (float*)(ws + 2 * AFRAG_BYTES +
                                 (size_t)SPLITK * BATCH * NCOL * 4);

    pack_patches_kernel<<<PACK_TOTAL / 256, 256, 0, stream>>>(x, Ahi, Alo);
    gemm_wmma_kernel<<<dim3(NCOL / 32, SPLITK), 32, 0, stream>>>(
        W, (const v16bf*)Ahi, (const v16bf*)Alo, out_part);
    bn_stats_kernel<<<OCH, 256, 0, stream>>>(out_part, bias, gamma, beta, scaleshift);
    bn_apply_kernel<<<(BATCH * NCOL) / 256, 256, 0, stream>>>(
        out_part, bias, scaleshift, out);
}
